// GlobalAggregator_30932354466500
// MI455X (gfx1250) — compile-verified, matrix operand order checked by reading, not run on hardware
//
#include <hip/hip_runtime.h>
#include <hip/hip_bf16.h>

// MI455X / gfx1250: wave32, bf16 WMMA 16x16x32, f32 accumulate.
typedef __attribute__((ext_vector_type(16))) __bf16 v16bf;
typedef __attribute__((ext_vector_type(8)))  __bf16 v8bf;
typedef __attribute__((ext_vector_type(4)))  __bf16 v4bf;
typedef __attribute__((ext_vector_type(8)))  float  v8f;

#define STR1 136   // padded stride (elems) for A / w1^T  (136*2B = 272B, 16B aligned rows)
#define STR2 264   // padded stride (elems) for A2 / [w3;w4]^T
#define SMEM_BYTES 87040  // max(192*STR1*2 + 128*STR1*2, 128*STR2*2 + 16*STR2*2)

// bf16 A/B fragment per ISA 16-bit layout: lane supplies row/col = lane%16,
// K values: VGPR0..3 <- K = h*8 + 0..7, VGPR4..7 <- K = 16 + h*8 + 0..7
// => two contiguous 16B LDS loads at +0 and +16 elems.
__device__ __forceinline__ v16bf lds_frag(const __bf16* p) {
  v8bf lo = *(const v8bf*)(p);
  v8bf hi = *(const v8bf*)(p + 16);
  return __builtin_shufflevector(lo, hi, 0,1,2,3,4,5,6,7,8,9,10,11,12,13,14,15);
}

__device__ __forceinline__ v8f wmma_bf16(v16bf a, v16bf b, v8f c) {
  // (neg_a, A, neg_b, B, c_mod, C, reuse_a, reuse_b)
  return __builtin_amdgcn_wmma_f32_16x16x32_bf16(false, a, false, b, (short)0, c, false, false);
}

__global__ __launch_bounds__(256)
void ga_kernel(const float* __restrict__ g_self,   // [P,128]
               const float* __restrict__ g_nb,     // [P,12,128]
               const float* __restrict__ g_nw,     // [P,12]
               const float* __restrict__ g_ext,    // [P,128]
               const float* __restrict__ g_w1,     // [129,128]
               const float* __restrict__ g_w2,     // [128]
               const float* __restrict__ g_w3,     // [128,128]
               const float* __restrict__ g_w4,     // [128,128]
               float* __restrict__ g_out,          // [P,128]
               int nPairs)
{
  __shared__ __attribute__((aligned(16))) char smem[SMEM_BYTES];
  __shared__ float s_logit[192];
  __shared__ float s_alpha[192];
  __shared__ float s_wrow[192];
  __shared__ float s_w2[128];
  __shared__ float s_w1last[128];

  // phase A layout
  __bf16* A    = (__bf16*)smem;              // 192 x STR1 : ext ⊙ neighbor (bf16)
  __bf16* W1T  = (__bf16*)(smem + 52224);    // 128 x STR1 : w1[:128]^T
  // phase B layout (aliases phase A, dead after stage 1)
  __bf16* W34T = (__bf16*)smem;              // 128 x STR2 : [w3;w4]^T
  __bf16* A2   = (__bf16*)(smem + 67584);    // 16  x STR2 : [self+neigh | self*neigh]

  const int tid  = threadIdx.x;
  const int wv   = tid >> 5;        // 8 waves
  const int lane = tid & 31;
  const int ln   = lane & 15;
  const int h    = lane >> 4;
  const long pairBase = (long)blockIdx.x * 16;

  // ---------------- phase 0: global -> LDS (f32 -> bf16) ----------------
  if (tid < 128) { s_w2[tid] = g_w2[tid]; s_w1last[tid] = g_w1[128 * 128 + tid]; }
  if (tid < 192) {
    s_logit[tid] = 0.0f;
    long p = pairBase + tid / 12; if (p >= nPairs) p = nPairs - 1;
    s_wrow[tid] = g_nw[p * 12 + (tid % 12)];
  }
  // w1[:128] transposed into W1T (B-matrix wants contiguous K per column)
  for (int i = 0; i < 64; ++i) {
    int e = tid + i * 256;                 // 16384 elems, e = k*128+n
    int k = e >> 7, n = e & 127;
    W1T[n * STR1 + k] = (__bf16)g_w1[e];
  }
  // A rows: row = p*12+k ; A[row,:] = extra[p,:] * neighbor[p,k,:]
  const float4* nb4p  = (const float4*)g_nb;
  const float4* ext4p = (const float4*)g_ext;
  for (int i = 0; i < 24; ++i) {
    int f = tid + i * 256;                 // 6144 float4s
    int row = f >> 5, d4 = f & 31;
    int p = row / 12, k = row - p * 12;
    long gp = pairBase + p; if (gp >= nPairs) gp = nPairs - 1;
    float4 nb = nb4p[(gp * 12 + k) * 32 + d4];
    float4 ex = ext4p[gp * 32 + d4];
    v4bf v;
    v[0] = (__bf16)(nb.x * ex.x);
    v[1] = (__bf16)(nb.y * ex.y);
    v[2] = (__bf16)(nb.z * ex.z);
    v[3] = (__bf16)(nb.w * ex.w);
    *(v4bf*)(A + row * STR1 + d4 * 4) = v;
  }
  __syncthreads();

  // ---- stage 1: C = A(192x128) @ w1[:128]; epilogue folds +w_k*w1[128,:],
  //      leaky_relu, dot with w2 -> logits (ds_add_f32 atomics) ----
  {
    const int nt   = wv;                   // wave owns one N-tile (16 cols)
    const int ncol = nt * 16 + ln;
    v16bf bfr[4];
    const __bf16* pb = W1T + ncol * STR1 + h * 8;
    #pragma unroll
    for (int kk = 0; kk < 4; ++kk) bfr[kk] = lds_frag(pb + kk * 32);
    const float bias = s_w1last[ncol];
    const float w2n  = s_w2[ncol];
    for (int mt = 0; mt < 12; ++mt) {
      v8f acc = {};
      const __bf16* pa = A + (mt * 16 + ln) * STR1 + h * 8;
      #pragma unroll
      for (int kk = 0; kk < 4; ++kk)
        acc = wmma_bf16(lds_frag(pa + kk * 32), bfr[kk], acc);
      #pragma unroll
      for (int r = 0; r < 8; ++r) {
        int row = mt * 16 + h * 8 + r;     // C layout: lane holds rows r, r+8
        float c = acc[r] + s_wrow[row] * bias;
        float a = (c > 0.0f) ? c : 0.2f * c;      // leaky_relu(0.2)
        atomicAdd(&s_logit[row], a * w2n);        // ds_add_f32
      }
    }
  }
  __syncthreads();

  // ---- softmax over K=12 per pair (lanes 0..15) || load [w3;w4]^T into LDS ----
  if (tid < 16) {
    float m = -3.4e38f;
    for (int k = 0; k < 12; ++k) { float l = s_logit[tid * 12 + k]; m = l > m ? l : m; }
    float ev[12], s = 0.0f;
    for (int k = 0; k < 12; ++k) { ev[k] = __expf(s_logit[tid * 12 + k] - m); s += ev[k]; }
    float inv = 1.0f / s;
    for (int k = 0; k < 12; ++k) s_alpha[tid * 12 + k] = ev[k] * inv;
  }
  for (int i = 0; i < 64; ++i) {
    int e = tid + i * 256; int k = e >> 7, n = e & 127;
    W34T[n * STR2 + k]       = (__bf16)g_w3[e];
    W34T[n * STR2 + 128 + k] = (__bf16)g_w4[e];
  }
  __syncthreads();

  // ---- stage 3: neigh = Σ_k α·neighbor (L2-hot re-read); build A2 ----
  for (int i = 0; i < 8; ++i) {
    int idx = tid + i * 256;               // 16 pairs x 128 dims
    int p = idx >> 7, dim = idx & 127;
    long gp = pairBase + p; if (gp >= nPairs) gp = nPairs - 1;
    float ng = 0.0f;
    const float* nbp = g_nb + gp * 12 * 128 + dim;
    #pragma unroll
    for (int k = 0; k < 12; ++k) ng += s_alpha[p * 12 + k] * nbp[k * 128];
    float sv = g_self[gp * 128 + dim];
    A2[p * STR2 + dim]       = (__bf16)(sv + ng);
    A2[p * STR2 + 128 + dim] = (__bf16)(sv * ng);
  }
  __syncthreads();

  // ---- stage 4: out = relu( A2(16x256) @ [w3;w4](256x128) ) ----
  {
    const int nt   = wv;
    const int ncol = nt * 16 + ln;
    v8f acc = {};
    const __bf16* pa = A2   + ln * STR2 + h * 8;
    const __bf16* pb = W34T + ncol * STR2 + h * 8;
    #pragma unroll
    for (int kk = 0; kk < 8; ++kk)
      acc = wmma_bf16(lds_frag(pa + kk * 32), lds_frag(pb + kk * 32), acc);
    #pragma unroll
    for (int r = 0; r < 8; ++r) {
      int p = h * 8 + r;
      long gp = pairBase + p;
      if (gp < nPairs) {
        float v = acc[r];
        g_out[gp * 128 + ncol] = v > 0.0f ? v : 0.0f;
      }
    }
  }
}

extern "C" void kernel_launch(void* const* d_in, const int* in_sizes, int n_in,
                              void* d_out, int out_size, void* d_ws, size_t ws_size,
                              hipStream_t stream) {
  const float* g_self = (const float*)d_in[0];   // self_vectors [B,N,128]
  const float* g_nb   = (const float*)d_in[1];   // neighbor_vector [B,N,12,128]
  const float* g_nw   = (const float*)d_in[2];   // neighbor_weight [B,N,12]
  const float* g_ext  = (const float*)d_in[3];   // extra_vector [B,N,128]
  const float* g_w1   = (const float*)d_in[4];   // [129,128]
  const float* g_w2   = (const float*)d_in[5];   // [128,1]
  const float* g_w3   = (const float*)d_in[6];   // [128,128]
  const float* g_w4   = (const float*)d_in[7];   // [128,128]
  float* g_out        = (float*)d_out;

  int nPairs = in_sizes[0] / 128;                // B*N = 51200
  int blocks = (nPairs + 15) / 16;               // 16 pairs per workgroup
  ga_kernel<<<blocks, 256, 0, stream>>>(g_self, g_nb, g_nw, g_ext,
                                        g_w1, g_w2, g_w3, g_w4, g_out, nPairs);
}
